// MambaStateSpaceModel_25795573580509
// MI455X (gfx1250) — matrix operand order repaired and checked by reading, not compiled, not run on hardware
//
#include <hip/hip_runtime.h>
#include <hip/hip_bf16.h>

// ---------------------------------------------------------------------------
// Mamba-style tanh RNN:  u = x@Wi^T+bi ; s_t = tanh(s_{t-1}@A_w^T + A_b + u_t)
//                        out = states@Wo^T + bo
// B=32, T=2048, D_IN=256, S=512, D_OUT=256. All WMMA bf16 (f32 accumulate).
// ---------------------------------------------------------------------------

#define Bb 32
#define Tt 2048
#define DIN 256
#define SS 512
#define DOUT 256

typedef __bf16 bf16_t;
typedef bf16_t v16bf __attribute__((ext_vector_type(16)));
typedef float  v8f   __attribute__((ext_vector_type(8)));
typedef unsigned short u16;
typedef unsigned int   u32;

// One 16-element bf16 fragment (A or B operand), 32 bytes.
union Frag {
    uint4  q[2];
    u32    u[8];
    bf16_t h[16];
    v16bf  v;
};

// Native converts (backend emits v_cvt_pk_bf16_f32 / v_cvt_bf16_f32).
static __device__ __forceinline__ u16 f2bf(float f) {
    union { bf16_t h; u16 b; } c; c.h = (bf16_t)f;
    return c.b;
}
static __device__ __forceinline__ u32 pack2(float a, float b) {
    union { bf16_t h[2]; u32 u; } p;
    p.h[0] = (bf16_t)a; p.h[1] = (bf16_t)b;
    return p.u;
}

// Native V_TANH_F32 on gfx1250 if the toolchain exposes it.
#if __has_builtin(__builtin_amdgcn_tanhf)
#define TANH(x) __builtin_amdgcn_tanhf(x)
#else
#define TANH(x) tanhf(x)
#endif

#define WMMA_BF16(A, Bm, C) \
    __builtin_amdgcn_wmma_f32_16x16x32_bf16(false, (A), false, (Bm), (short)0, (C), false, false)

// ---------------------------------------------------------------------------
// Kernel 0: convert A_w / Wi / Wo to bf16 once (pairwise -> v_cvt_pk_bf16_f32)
// ---------------------------------------------------------------------------
__global__ void cvt_weights(const float* __restrict__ Aw, const float* __restrict__ Wi,
                            const float* __restrict__ Wo,
                            u32* __restrict__ Awb, u32* __restrict__ Wib, u32* __restrict__ Wob) {
    const int n1 = SS * SS / 2, n2 = SS * DIN / 2, n3 = DOUT * SS / 2;
    for (int i = blockIdx.x * blockDim.x + threadIdx.x; i < n1 + n2 + n3;
         i += gridDim.x * blockDim.x) {
        if (i < n1) {
            float2 f = ((const float2*)Aw)[i];
            Awb[i] = pack2(f.x, f.y);
        } else if (i < n1 + n2) {
            float2 f = ((const float2*)Wi)[i - n1];
            Wib[i - n1] = pack2(f.x, f.y);
        } else {
            float2 f = ((const float2*)Wo)[i - n1 - n2];
            Wob[i - n1 - n2] = pack2(f.x, f.y);
        }
    }
}

// ---------------------------------------------------------------------------
// Kernel 1: input projection u[t][b][s] = x[b][t][:] . Wi[s][:] + bi[s]
// 256 threads = 8 waves; block = one 16-row M-tile (rows bt = b*T+t).
// The 16x256 x tile is converted to bf16 ONCE into LDS, then all 8 waves
// (one 64-col N-group each) read A-fragments with ds_load_b128.
// ---------------------------------------------------------------------------
#define XSP 264  // LDS row stride (bf16) for the x tile: 16B-aligned rows, no conflicts

__global__ __launch_bounds__(256) void inproj(const float* __restrict__ x,
                                              const u16* __restrict__ Wib,
                                              const float* __restrict__ bi,
                                              float* __restrict__ u) {
    __shared__ u16 xs[16 * XSP];
    const int tid  = threadIdx.x;
    const int lane = tid & 31;
    const int wave = tid >> 5;
    const int lr = lane & 15, hi = lane >> 4;
    const int bt0 = blockIdx.x * 16;
    const int n0  = wave * 64;

    // ---- stage x tile (16 rows x 256 cols) as bf16 into LDS ----
    {
        const int row  = tid >> 4;          // 0..15
        const int col0 = (tid & 15) * 16;   // 0..240
        const float4* src = (const float4*)(x + ((size_t)(bt0 + row)) * DIN + col0);
        Frag tmpa;
        float4 f0 = src[0], f1 = src[1], f2 = src[2], f3 = src[3];
        tmpa.u[0] = pack2(f0.x, f0.y); tmpa.u[1] = pack2(f0.z, f0.w);
        tmpa.u[2] = pack2(f1.x, f1.y); tmpa.u[3] = pack2(f1.z, f1.w);
        tmpa.u[4] = pack2(f2.x, f2.y); tmpa.u[5] = pack2(f2.z, f2.w);
        tmpa.u[6] = pack2(f3.x, f3.y); tmpa.u[7] = pack2(f3.z, f3.w);
        uint4* dst = (uint4*)&xs[row * XSP + col0];
        dst[0] = tmpa.q[0]; dst[1] = tmpa.q[1];
    }
    __syncthreads();

    v8f acc[4] = {};
#pragma unroll
    for (int kc = 0; kc < DIN / 32; ++kc) {
        // A fragment straight from LDS (two b128 loads, conflict-free)
        Frag a;
        const int kb = kc * 32 + hi * 16;
        a.q[0] = *(const uint4*)&xs[lr * XSP + kb];
        a.q[1] = *(const uint4*)&xs[lr * XSP + kb + 8];
        const int kbb = kc * 32 + hi * 16;
#pragma unroll
        for (int j = 0; j < 4; ++j) {
            const int n = n0 + j * 16 + lr;
            const uint4* bp = (const uint4*)(Wib + (size_t)n * DIN + kbb);
            Frag b; b.q[0] = bp[0]; b.q[1] = bp[1];
            acc[j] = WMMA_BF16(a.v, b.v, acc[j]);
        }
    }
    // epilogue: scatter to u in [T][B][S] (t-major so the scan reads contiguously)
#pragma unroll
    for (int j = 0; j < 4; ++j) {
        const int n = n0 + j * 16 + lr;
        const float bin = bi[n];
#pragma unroll
        for (int r = 0; r < 8; ++r) {
            const int m  = r + hi * 8;
            const int bt = bt0 + m;
            const int b  = bt >> 11;        // /T
            const int t  = bt & (Tt - 1);
            u[((size_t)t * Bb + b) * SS + n] = acc[j][r] + bin;
        }
    }
}

// ---------------------------------------------------------------------------
// Kernel 2: sequential recurrence. ONE workgroup, 1024 threads = 32 waves.
// State (32x512 bf16) lives in LDS (row stride 520 -> b128 reads conflict-free).
// Wave w owns N-tile w (16 cols); its 16x512 slice of A_w stays in VGPRs
// (Frag B[16] = 128 VGPRs) for the whole 2048-step scan -> zero weight traffic.
// u_t is loaded at the TOP of each step (overlaps the 32 WMMAs), and u_{t+1}
// is prefetched; tanh uses the native V_TANH_F32 transcendental when available.
// ---------------------------------------------------------------------------
#define STP 520  // padded LDS row stride (bf16 elements); row*STP*2 is 16B-aligned

__global__ __launch_bounds__(1024, 1) void recurrence(const u16* __restrict__ Awb,
                                                      const float* __restrict__ Ab,
                                                      const float* __restrict__ u,
                                                      u16* __restrict__ states) {
    __shared__ u16 st[Bb * STP];
    const int tid  = threadIdx.x;
    const int lane = tid & 31, wave = tid >> 5;
    const int lr = lane & 15, hi = lane >> 4;
    const int n = wave * 16 + lr;                // output column owned by this lane

    for (int i = tid; i < Bb * STP; i += 1024) st[i] = 0;

    // hoist this wave's A_w slice into registers (B fragments, all 16 K-chunks)
    Frag B[16];
#pragma unroll
    for (int kc = 0; kc < 16; ++kc) {
        const uint4* p = (const uint4*)(Awb + (size_t)n * SS + kc * 32 + hi * 16);
        B[kc].q[0] = p[0]; B[kc].q[1] = p[1];
    }
    const float ab = Ab[n];
    __syncthreads();

    for (int t = 0; t < Tt; ++t) {
        const float* urow = u + (size_t)t * Bb * SS;

        // issue u_t loads first: they resolve while the WMMAs run
        float uv0[8], uv1[8];
#pragma unroll
        for (int r = 0; r < 8; ++r) {
            const int m0 = r + hi * 8;
            uv0[r] = urow[m0 * SS + n];
            uv1[r] = urow[(16 + m0) * SS + n];
        }
        // prefetch next step's u row (1024 threads x 64B == the full 64KB row)
        if (t + 1 < Tt)
            __builtin_prefetch(u + (size_t)(t + 1) * Bb * SS + tid * 16, 0, 0);

        v8f acc0 = {}, acc1 = {};
#pragma unroll
        for (int kc = 0; kc < 16; ++kc) {
            const int kb = kc * 32 + hi * 16;
            Frag a0, a1;
            a0.q[0] = *(const uint4*)&st[lr * STP + kb];
            a0.q[1] = *(const uint4*)&st[lr * STP + kb + 8];
            a1.q[0] = *(const uint4*)&st[(16 + lr) * STP + kb];
            a1.q[1] = *(const uint4*)&st[(16 + lr) * STP + kb + 8];
            acc0 = WMMA_BF16(a0.v, B[kc].v, acc0);
            acc1 = WMMA_BF16(a1.v, B[kc].v, acc1);
        }
        __syncthreads();                         // all reads of s_{t-1} done
#pragma unroll
        for (int r = 0; r < 8; ++r) {
            const int m0 = r + hi * 8;
            const int m1 = 16 + m0;
            const float s0 = TANH(acc0[r] + ab + uv0[r]);
            const float s1 = TANH(acc1[r] + ab + uv1[r]);
            const u16 h0 = f2bf(s0), h1 = f2bf(s1);
            st[m0 * STP + n] = h0;
            st[m1 * STP + n] = h1;
            states[((size_t)t * Bb + m0) * SS + n] = h0;
            states[((size_t)t * Bb + m1) * SS + n] = h1;
        }
        __syncthreads();                         // s_t visible to everyone
    }
}

// ---------------------------------------------------------------------------
// Kernel 3: output projection out[b][t][o] = states[t*32+b][:] . Wo[o][:] + bo
// 256 threads = 8 waves; block covers 2 M-tiles x 4 N-groups of 64.
// ---------------------------------------------------------------------------
__global__ __launch_bounds__(256) void outproj(const u16* __restrict__ states,
                                               const u16* __restrict__ Wob,
                                               const float* __restrict__ bo,
                                               float* __restrict__ out) {
    const int lane = threadIdx.x & 31;
    const int wave = threadIdx.x >> 5;
    const int lr = lane & 15, hi = lane >> 4;
    const int mt = blockIdx.x * 2 + (wave >> 2);
    const int n0 = (wave & 3) * 64;
    const int r0 = mt * 16;

    v8f acc[4] = {};
    const u16* arow = states + (size_t)(r0 + lr) * SS;

#pragma unroll 4
    for (int kc = 0; kc < SS / 32; ++kc) {
        Frag a;
        const int kb = kc * 32 + hi * 16;
        a.q[0] = *(const uint4*)(arow + kb);
        a.q[1] = *(const uint4*)(arow + kb + 8);
        const int kbb = kc * 32 + hi * 16;
#pragma unroll
        for (int j = 0; j < 4; ++j) {
            const int o = n0 + j * 16 + lr;
            const uint4* bp = (const uint4*)(Wob + (size_t)o * SS + kbb);
            Frag b; b.q[0] = bp[0]; b.q[1] = bp[1];
            acc[j] = WMMA_BF16(a.v, b.v, acc[j]);
        }
    }
#pragma unroll
    for (int j = 0; j < 4; ++j) {
        const int o = n0 + j * 16 + lr;
        const float bon = bo[o];
#pragma unroll
        for (int r = 0; r < 8; ++r) {
            const int row = r0 + r + hi * 8;   // = t*32 + b
            const int t = row >> 5, b = row & 31;
            out[((size_t)b * Tt + t) * DOUT + o] = acc[j][r] + bon;
        }
    }
}

// ---------------------------------------------------------------------------
// Launch. Workspace layout (needs ~193 MB):
//   [0, 128MB)        u        f32 [T][B][S]
//   [128MB, 192MB)    states   bf16 [T][B][S]
//   [192MB, ...)      A_w bf16 (512KB), Wi bf16 (256KB), Wo bf16 (256KB)
// ---------------------------------------------------------------------------
extern "C" void kernel_launch(void* const* d_in, const int* in_sizes, int n_in,
                              void* d_out, int out_size, void* d_ws, size_t ws_size,
                              hipStream_t stream) {
    const float* x  = (const float*)d_in[0];
    const float* Aw = (const float*)d_in[1];
    const float* Ab = (const float*)d_in[2];
    const float* Wi = (const float*)d_in[3];
    const float* bi = (const float*)d_in[4];
    const float* Wo = (const float*)d_in[5];
    const float* bo = (const float*)d_in[6];
    float* out = (float*)d_out;

    char* ws = (char*)d_ws;
    const size_t U_BYTES  = (size_t)Tt * Bb * SS * sizeof(float);   // 128 MB
    const size_t ST_BYTES = (size_t)Tt * Bb * SS * sizeof(u16);     //  64 MB
    float* u      = (float*)ws;
    u16*   states = (u16*)(ws + U_BYTES);
    u16*   Awb    = (u16*)(ws + U_BYTES + ST_BYTES);
    u16*   Wib    = Awb + (size_t)SS * SS;
    u16*   Wob    = Wib + (size_t)SS * DIN;

    cvt_weights<<<512, 256, 0, stream>>>(Aw, Wi, Wo, (u32*)Awb, (u32*)Wib, (u32*)Wob);
    inproj<<<(Bb * Tt) / 16, 256, 0, stream>>>(x, Wib, bi, u);
    recurrence<<<1, 1024, 0, stream>>>(Awb, Ab, u, states);
    outproj<<<(Bb * Tt) / 32, 256, 0, stream>>>(states, Wob, bo, out);
}
